// SwinBlock_51479478010539
// MI455X (gfx1250) — matrix operand order repaired and checked by reading, not compiled, not run on hardware
//
#include <hip/hip_runtime.h>
#include <hip/hip_bf16.h>
#include <math.h>

// ---------------------------------------------------------------------------
// Types for CDNA5 WMMA
// ---------------------------------------------------------------------------
typedef __attribute__((ext_vector_type(16))) __bf16 v16bf;
typedef __attribute__((ext_vector_type(8)))  __bf16 v8bf;
typedef __attribute__((ext_vector_type(8)))  float  v8f;

static __device__ __forceinline__ __bf16 f2bf(float f) {
    unsigned u = __builtin_bit_cast(unsigned, f);
    u += 0x7FFFu + ((u >> 16) & 1u);           // round-to-nearest-even
    unsigned short s = (unsigned short)(u >> 16);
    return __builtin_bit_cast(__bf16, s);
}
static __device__ __forceinline__ __bf16 bf_zero() {
    return __builtin_bit_cast(__bf16, (unsigned short)0);
}

static __device__ __forceinline__ v8f wmma_bf16(v16bf a, v16bf b, v8f c) {
    // D = A(16x32 bf16) * B(32x16 bf16) + C(16x16 f32)
    return __builtin_amdgcn_wmma_f32_16x16x32_bf16(
        /*neg_a=*/false, a, /*neg_b=*/false, b,
        /*c_mod=*/(short)0, c, /*reuse_a=*/false, /*reuse_b=*/false);
}

// Build an A-fragment (16x32, 16-bit) for one lane.
// rowk points at the 32-element K-slice of this lane's row.
// Lane half 0 holds K{0..15}; half 1 holds K{16..31}.
static __device__ __forceinline__ v16bf make_afrag(const __bf16* rowk, int half) {
    return *(const v16bf*)(rowk + half * 16);
}

// CDNA5 async global -> LDS copy (16B per lane), tracked by ASYNCcnt.
static __device__ __forceinline__ void async_copy_b128(unsigned lds_addr,
                                                       const __bf16* gaddr) {
    asm volatile("global_load_async_to_lds_b128 %0, %1, off"
                 :: "v"(lds_addr), "v"((unsigned long long)(uintptr_t)gaddr)
                 : "memory");
}
static __device__ __forceinline__ void wait_asynccnt0() {
    asm volatile("s_wait_asynccnt 0x0" ::: "memory");
}

// ---------------------------------------------------------------------------
// Problem constants (hardcoded from reference: B=16, H=W=56, C=512)
// ---------------------------------------------------------------------------
#define CB    16
#define CH    56
#define CL    3136
#define CC    512
#define CNH   16
#define CDH   32
#define CWS   7
#define CN    49
#define CNW   64
#define CBW   1024        // B * nW
#define CM    50176       // total rows = B*nW*N = B*L
#define CMLP  2048

// ---------------------------------------------------------------------------
// K1: f32 -> bf16 convert
// ---------------------------------------------------------------------------
__global__ void k_tobf16(const float* __restrict__ s, __bf16* __restrict__ d, int n) {
    int i = blockIdx.x * 256 + threadIdx.x;
    if (i < n) d[i] = f2bf(s[i]);
}

// ---------------------------------------------------------------------------
// K2: mod = silu(emb) @ adaLN_w + adaLN_b   -> [16, 3072] f32
// ---------------------------------------------------------------------------
__global__ void k_adaln(const float* __restrict__ emb, const float* __restrict__ w,
                        const float* __restrict__ bias, float* __restrict__ mod) {
    int o = blockIdx.x * 256 + threadIdx.x;       // 16*3072 = 49152 outputs
    int b = o / 3072, c = o - b * 3072;
    float acc = bias[c];
    const float* er = emb + b * 512;
    for (int k = 0; k < 512; ++k) {
        float e = er[k];
        float si = e / (1.0f + __expf(-e));
        acc += si * w[(size_t)k * 3072 + c];
    }
    mod[o] = acc;
}

// ---------------------------------------------------------------------------
// K3/K6: LN + AdaLN modulate (+ optional shift-roll + window partition), bf16 out
// One wave per row (wave32), 16 columns per lane.
// ---------------------------------------------------------------------------
__global__ void __launch_bounds__(256)
k_ln_mod(const float* __restrict__ x, const float* __restrict__ mod,
         __bf16* __restrict__ out, int windowed, int offSh, int offSc) {
    int row  = blockIdx.x * 8 + (threadIdx.x >> 5);
    int lane = threadIdx.x & 31;

    int b;
    const float* xr;
    if (windowed) {
        int b_ = row / CN, n = row - b_ * CN;
        b = b_ >> 6;
        int wi = b_ & 63, wh = wi >> 3, ww = wi & 7;
        int i = n / CWS, j = n - CWS * i;
        int gh = (wh * CWS + i + 3) % CH;       // roll(-shift) source index
        int gw = (ww * CWS + j + 3) % CH;
        xr = x + ((size_t)b * CL + gh * CH + gw) * CC;
    } else {
        b = row / CL;
        xr = x + (size_t)row * CC;
    }

    float v[16], s = 0.f, ss = 0.f;
#pragma unroll
    for (int j = 0; j < 16; ++j) {
        v[j] = xr[lane + 32 * j];
        s += v[j]; ss += v[j] * v[j];
    }
#pragma unroll
    for (int d = 1; d < 32; d <<= 1) {
        s  += __shfl_xor(s,  d, 32);
        ss += __shfl_xor(ss, d, 32);
    }
    float mu   = s * (1.0f / 512.0f);
    float var  = ss * (1.0f / 512.0f) - mu * mu;
    float rstd = rsqrtf(var + 1e-6f);

    const float* mb = mod + (size_t)b * 3072;
    __bf16* orow = out + (size_t)row * CC;
#pragma unroll
    for (int j = 0; j < 16; ++j) {
        int c = lane + 32 * j;
        float val = (v[j] - mu) * rstd * (1.0f + mb[offSc + c]) + mb[offSh + c];
        orow[c] = f2bf(val);
    }
}

// ---------------------------------------------------------------------------
// GEMM epilogues
//   0: qkv  (+bias, scatter to [s][b_][h][n][d] bf16)
//   1: proj (+bias, window-reverse+unshift, x1 = x + g_sa*val)
//   2: ff1  (+bias, exact GELU, bf16)
//   3: ff2  (+bias, out = x1 + g_ff*val, f32)
// ---------------------------------------------------------------------------
template <int EPI>
static __device__ __forceinline__ void epi_store(
    float val, int r, int c,
    const float* __restrict__ bias, const float* __restrict__ mod,
    const float* __restrict__ resid, float* __restrict__ outf,
    __bf16* __restrict__ outb) {
    val += bias[c];
    if constexpr (EPI == 0) {
        int s = c >> 9, hh = (c >> 5) & 15, d = c & 31;
        int b_ = r / CN, n = r - b_ * CN;
        outb[((((size_t)s * CBW + b_) * CNH + hh) * CN + n) * CDH + d] = f2bf(val);
    } else if constexpr (EPI == 1) {
        int b_ = r / CN, n = r - b_ * CN;
        int b = b_ >> 6, wi = b_ & 63;
        int i = n / CWS, j = n - CWS * i;
        int gh = ((wi >> 3) * CWS + i + 3) % CH;
        int gw = ((wi & 7)  * CWS + j + 3) % CH;
        size_t idx = ((size_t)b * CL + gh * CH + gw) * CC + c;
        outf[idx] = resid[idx] + mod[(size_t)b * 3072 + 1024 + c] * val;
    } else if constexpr (EPI == 2) {
        float g = 0.5f * val * (1.0f + erff(val * 0.70710678118654752f));
        outb[(size_t)r * CMLP + c] = f2bf(g);
    } else {
        int b = r / CL;
        size_t idx = (size_t)r * CC + c;
        outf[idx] = resid[idx] + mod[(size_t)b * 3072 + 2560 + c] * val;
    }
}

// ---------------------------------------------------------------------------
// K4: bf16 WMMA GEMM, C = A[M,K] * B[K,N]
// 128x64 tile per 256-thread (8-wave) block; each wave owns a 32x32 subtile.
// A tiles move via async global->LDS (ASYNCcnt), double-buffered;
// B tiles are register-staged + transposed, overlapped with compute.
// One barrier per k-step.
// ---------------------------------------------------------------------------
template <int EPI>
__global__ void __launch_bounds__(256)
k_wgemm(const __bf16* __restrict__ A, const __bf16* __restrict__ Bw,
        int M, int N, int K,
        const float* __restrict__ bias, const float* __restrict__ mod,
        const float* __restrict__ resid, float* __restrict__ outf,
        __bf16* __restrict__ outb) {
    __shared__ __align__(16) __bf16 sa [2][128 * 32];  // A tiles, row-major
    __shared__ __align__(16) __bf16 sbt[2][64 * 32];   // B tiles, transposed [n][k]

    const int tid  = threadIdx.x;
    const int lane = tid & 31;
    const int w    = tid >> 5;
    const int m0   = blockIdx.x * 128;
    const int n0   = blockIdx.y * 64;
    const int wm   = (w >> 1) * 32;
    const int wn   = (w & 1)  * 32;
    const int rl   = lane & 15, half = lane >> 4;

    // A async-copy mapping: 512 x 16B chunks, 2 per thread
    const int ac0 = tid * 2, ac1 = tid * 2 + 1;
    const int ar0 = ac0 >> 2, ap0 = ac0 & 3;           // row, 8-elem part
    const int ar1 = ac1 >> 2, ap1 = ac1 & 3;
    // B loader mapping: 32 k-rows x 64 n, 8 threads per k-row, 8 elems each
    const int bk = tid >> 3, bq = tid & 7;

    v8f acc00 = {}, acc01 = {}, acc10 = {}, acc11 = {};

    const int nk = K >> 5;
    union Bu { uint4 u; __bf16 e[8]; };

    // ---- prologue: stage tile 0 ----
    async_copy_b128((unsigned)(uintptr_t)&sa[0][ar0 * 32 + ap0 * 8],
                    A + (size_t)(m0 + ar0) * K + ap0 * 8);
    async_copy_b128((unsigned)(uintptr_t)&sa[0][ar1 * 32 + ap1 * 8],
                    A + (size_t)(m0 + ar1) * K + ap1 * 8);
    {
        Bu t; t.u = *(const uint4*)(Bw + (size_t)bk * N + n0 + bq * 8);
#pragma unroll
        for (int j = 0; j < 8; ++j) sbt[0][(bq * 8 + j) * 32 + bk] = t.e[j];
    }

    for (int i = 0; i < nk; ++i) {
        const int p = i & 1;
        wait_asynccnt0();            // own async A copies for tile i complete
        __syncthreads();             // all waves' tile-i data visible; prev compute done

        Bu t;
        const bool more = (i + 1) < nk;
        if (more) {
            const int k1 = (i + 1) << 5;
            // kick off next A tile (async, writes buffer ~p)
            async_copy_b128((unsigned)(uintptr_t)&sa[p ^ 1][ar0 * 32 + ap0 * 8],
                            A + (size_t)(m0 + ar0) * K + k1 + ap0 * 8);
            async_copy_b128((unsigned)(uintptr_t)&sa[p ^ 1][ar1 * 32 + ap1 * 8],
                            A + (size_t)(m0 + ar1) * K + k1 + ap1 * 8);
            // next B tile into registers (wait folded after compute)
            t.u = *(const uint4*)(Bw + (size_t)(k1 + bk) * N + n0 + bq * 8);
        }

        // ---- compute tile i ----
        v16bf a0 = make_afrag(&sa[p][(wm + rl) * 32], half);
        v16bf a1 = make_afrag(&sa[p][(wm + 16 + rl) * 32], half);
        v16bf b0 = *(const v16bf*)&sbt[p][(wn + rl) * 32 + half * 16];
        v16bf b1 = *(const v16bf*)&sbt[p][(wn + 16 + rl) * 32 + half * 16];
        acc00 = wmma_bf16(a0, b0, acc00);
        acc01 = wmma_bf16(a0, b1, acc01);
        acc10 = wmma_bf16(a1, b0, acc10);
        acc11 = wmma_bf16(a1, b1, acc11);

        if (more) {
#pragma unroll
            for (int j = 0; j < 8; ++j)
                sbt[p ^ 1][(bq * 8 + j) * 32 + bk] = t.e[j];
        }
    }

#pragma unroll
    for (int vr = 0; vr < 8; ++vr) {
        int rb = m0 + wm + vr + 8 * half;
        int cb = n0 + wn + rl;
        epi_store<EPI>(acc00[vr], rb,      cb,      bias, mod, resid, outf, outb);
        epi_store<EPI>(acc01[vr], rb,      cb + 16, bias, mod, resid, outf, outb);
        epi_store<EPI>(acc10[vr], rb + 16, cb,      bias, mod, resid, outf, outb);
        epi_store<EPI>(acc11[vr], rb + 16, cb + 16, bias, mod, resid, outf, outb);
    }
}

// ---------------------------------------------------------------------------
// K5: windowed attention, one block (4 waves) per (window b_, head h)
// ---------------------------------------------------------------------------
static __device__ __forceinline__ int reg3(int g) {
    return g < 49 ? 0 : (g < 53 ? 1 : 2);
}

__global__ void __launch_bounds__(128)
k_attn(const __bf16* __restrict__ qkv, const float* __restrict__ table,
       __bf16* __restrict__ out) {
    __shared__ __align__(16) __bf16 sq [64 * 32];   // q rows padded to 64
    __shared__ __align__(16) __bf16 sk [64 * 32];   // k rows padded to 64
    __shared__ __align__(16) __bf16 svt[32 * 64];   // v transposed [d][n]
    __shared__ __align__(16) __bf16 sp [4][16 * 64];// per-wave softmax(P) tile

    const int b_ = blockIdx.x, h = blockIdx.y;
    const int tid = threadIdx.x, lane = tid & 31, w = tid >> 5;
    const int rl = lane & 15, half = lane >> 4;

    const size_t sstride = (size_t)CBW * CNH * CN * CDH;
    const __bf16* gq = qkv + (size_t)(b_ * CNH + h) * (CN * CDH);
    const __bf16* gk = gq + sstride;
    const __bf16* gv = gq + 2 * sstride;

    const __bf16 z = bf_zero();
    for (int i = tid; i < 2048; i += 128) { sq[i] = z; sk[i] = z; svt[i] = z; }
    __syncthreads();
    for (int i = tid; i < 196; i += 128) {                 // 1568 bf16 = 196 uint4
        ((uint4*)sq)[i] = ((const uint4*)gq)[i];
        ((uint4*)sk)[i] = ((const uint4*)gk)[i];
    }
    for (int e = tid; e < CN * CDH; e += 128) {            // v -> transposed
        int n = e >> 5, d = e & 31;
        svt[d * 64 + n] = gv[e];
    }
    __syncthreads();

    // ---- S = q k^T (each wave: 16 rows x 64 cols, dh=32 = one k-step) ----
    v16bf aq = make_afrag(&sq[(w * 16 + rl) * 32], half);
    v8f accS[4];
#pragma unroll
    for (int t = 0; t < 4; ++t) {
        accS[t] = (v8f){};
        v16bf bk_ = *(const v16bf*)&sk[(t * 16 + rl) * 32 + half * 16];
        accS[t] = wmma_bf16(aq, bk_, accS[t]);
    }

    // ---- scale + rel-pos bias + shift mask ----
    const float scale = 0.17677669529663688f;              // 1/sqrt(32)
    const int wi = b_ & 63, wh = wi >> 3, ww = wi & 7;
#pragma unroll
    for (int t = 0; t < 4; ++t) {
#pragma unroll
        for (int vr = 0; vr < 8; ++vr) {
            int n = w * 16 + vr + 8 * half;                // query position
            int m = t * 16 + rl;                           // key position
            float sv = accS[t][vr] * scale;
            if (m >= CN) {
                sv = -3.0e38f;
            } else if (n < CN) {
                int i1 = n / 7, j1 = n - 7 * i1, i2 = m / 7, j2 = m - 7 * i2;
                sv += table[((i1 - i2 + 6) * 13 + (j1 - j2 + 6)) * CNH + h];
                int r1 = reg3(wh * 7 + i1) * 3 + reg3(ww * 7 + j1);
                int r2 = reg3(wh * 7 + i2) * 3 + reg3(ww * 7 + j2);
                if (r1 != r2) sv -= 100.0f;
            }
            accS[t][vr] = sv;
        }
    }

    // ---- row softmax (row lives in one 16-lane half) ----
#pragma unroll
    for (int vr = 0; vr < 8; ++vr) {
        float mx = accS[0][vr];
        mx = fmaxf(mx, accS[1][vr]); mx = fmaxf(mx, accS[2][vr]); mx = fmaxf(mx, accS[3][vr]);
#pragma unroll
        for (int d = 1; d < 16; d <<= 1) mx = fmaxf(mx, __shfl_xor(mx, d, 32));
        float s = 0.f;
#pragma unroll
        for (int t = 0; t < 4; ++t) {
            float e = __expf(accS[t][vr] - mx);
            accS[t][vr] = e; s += e;
        }
#pragma unroll
        for (int d = 1; d < 16; d <<= 1) s += __shfl_xor(s, d, 32);
        float inv = 1.0f / s;
#pragma unroll
        for (int t = 0; t < 4; ++t) {
            accS[t][vr] *= inv;
            sp[w][(vr + 8 * half) * 64 + t * 16 + rl] = f2bf(accS[t][vr]);
        }
    }
    __syncthreads();

    // ---- O = P V (K=64 -> 2 k-steps; N=32 -> 2 tiles) ----
    v8f accO0 = {}, accO1 = {};
#pragma unroll
    for (int ks = 0; ks < 2; ++ks) {
        v16bf pa = make_afrag(&sp[w][rl * 64 + ks * 32], half);
        v16bf v0 = *(const v16bf*)&svt[(rl)      * 64 + ks * 32 + half * 16];
        v16bf v1 = *(const v16bf*)&svt[(16 + rl) * 64 + ks * 32 + half * 16];
        accO0 = wmma_bf16(pa, v0, accO0);
        accO1 = wmma_bf16(pa, v1, accO1);
    }

#pragma unroll
    for (int vr = 0; vr < 8; ++vr) {
        int n = w * 16 + vr + 8 * half;
        if (n < CN) {
            size_t r = (size_t)b_ * CN + n;
            out[r * CC + h * CDH + rl]      = f2bf(accO0[vr]);
            out[r * CC + h * CDH + 16 + rl] = f2bf(accO1[vr]);
        }
    }
}

// ---------------------------------------------------------------------------
// Host-side orchestration
// ---------------------------------------------------------------------------
extern "C" void kernel_launch(void* const* d_in, const int* in_sizes, int n_in,
                              void* d_out, int out_size, void* d_ws, size_t ws_size,
                              hipStream_t stream) {
    const float* x      = (const float*)d_in[0];
    const float* emb    = (const float*)d_in[1];
    const float* aw     = (const float*)d_in[2];
    const float* ab     = (const float*)d_in[3];
    const float* qkv_w  = (const float*)d_in[4];
    const float* qkv_b  = (const float*)d_in[5];
    const float* proj_w = (const float*)d_in[6];
    const float* proj_b = (const float*)d_in[7];
    const float* rtab   = (const float*)d_in[8];
    const float* ff_w1  = (const float*)d_in[9];
    const float* ff_b1  = (const float*)d_in[10];
    const float* ff_w2  = (const float*)d_in[11];
    const float* ff_b2  = (const float*)d_in[12];

    char* ws = (char*)d_ws;
    // workspace layout (bytes)
    float*  mod  = (float*)(ws + 0);                     //  786,432
    __bf16* Wq   = (__bf16*)(ws + 786432);               //  786,432 el
    __bf16* Wp   = Wq + 786432;                          //  262,144 el
    __bf16* W1   = Wp + 262144;                          // 1,048,576 el
    __bf16* W2   = W1 + 1048576;                         // 1,048,576 el
    __bf16* hwin = (__bf16*)(ws + 7077888);              // 25,690,112 el
    __bf16* qkv  = (__bf16*)(ws + 58458112);             // 77,070,336 el
    float*  x1   = (float*)(ws + 212598784);             // 25,690,112 f32
    __bf16* h2   = (__bf16*)(ws + 315359232);            // 25,690,112 el
    __bf16* attno = hwin;                                // reuse (hwin dead after QKV GEMM)
    __bf16* ff1   = hwin;                                // reuse hwin+qkv (205 MB contiguous)

    // 1) weights -> bf16
    k_tobf16<<<(786432 + 255) / 256, 256, 0, stream>>>(qkv_w, Wq, 786432);
    k_tobf16<<<(262144 + 255) / 256, 256, 0, stream>>>(proj_w, Wp, 262144);
    k_tobf16<<<(1048576 + 255) / 256, 256, 0, stream>>>(ff_w1, W1, 1048576);
    k_tobf16<<<(1048576 + 255) / 256, 256, 0, stream>>>(ff_w2, W2, 1048576);

    // 2) AdaLN modulation vector
    k_adaln<<<192, 256, 0, stream>>>(emb, aw, ab, mod);

    // 3) LN + modulate + shift + window partition -> hwin (bf16)
    k_ln_mod<<<CM / 8, 256, 0, stream>>>(x, mod, hwin, 1, 0, 512);

    // 4) QKV GEMM: [50176,512] x [512,1536]
    k_wgemm<0><<<dim3(CM / 128, 1536 / 64), 256, 0, stream>>>(
        hwin, Wq, CM, 1536, 512, qkv_b, nullptr, nullptr, nullptr, qkv);

    // 5) windowed attention
    k_attn<<<dim3(CBW, CNH), 128, 0, stream>>>(qkv, rtab, attno);

    // 6) proj GEMM + window-reverse + unshift + gated residual -> x1 (f32)
    k_wgemm<1><<<dim3(CM / 128, 512 / 64), 256, 0, stream>>>(
        attno, Wp, CM, 512, 512, proj_b, mod, x, x1, nullptr);

    // 7) LN + modulate (FF branch) -> h2 (bf16)
    k_ln_mod<<<CM / 8, 256, 0, stream>>>(x1, mod, h2, 0, 1536, 2048);

    // 8) FF1 GEMM + GELU: [50176,512] x [512,2048]
    k_wgemm<2><<<dim3(CM / 128, CMLP / 64), 256, 0, stream>>>(
        h2, W1, CM, CMLP, 512, ff_b1, nullptr, nullptr, nullptr, ff1);

    // 9) FF2 GEMM + gated residual -> out (f32)
    k_wgemm<3><<<dim3(CM / 128, 512 / 64), 256, 0, stream>>>(
        ff1, W2, CM, 512, CMLP, ff_b2, mod, x1, (float*)d_out, nullptr);
}
